// Block_43224550867974
// MI455X (gfx1250) — compile-verified, matrix operand order checked by reading, not run on hardware
//
#include <hip/hip_runtime.h>
#include <math.h>

typedef __bf16 bf16_t;
typedef __attribute__((ext_vector_type(16))) __bf16 v16bf;
typedef __attribute__((ext_vector_type(8)))  float  v8f;
typedef unsigned int u32x4 __attribute__((ext_vector_type(4)));
typedef int i32x4 __attribute__((ext_vector_type(4)));

union AFrag { v16bf v; u32x4 q[2]; };

#if __has_builtin(__builtin_amdgcn_global_load_async_to_lds_b128) && \
    __has_builtin(__builtin_amdgcn_s_wait_asynccnt)
#define HAS_ASYNC_LDS 1
typedef __attribute__((address_space(1))) i32x4* glds_gptr;
typedef __attribute__((address_space(3))) i32x4* glds_lptr;
#else
#define HAS_ASYNC_LDS 0
#endif

__device__ __forceinline__ void async_copy16(void* lds, const void* g) {
#if HAS_ASYNC_LDS
    __builtin_amdgcn_global_load_async_to_lds_b128(
        (glds_gptr)(g), (glds_lptr)(lds), 0, 0);
#endif
}
__device__ __forceinline__ void async_wait0() {
#if HAS_ASYNC_LDS
    __builtin_amdgcn_s_wait_asynccnt(0);
#endif
}

// ---------------------------------------------------------------------------
// Weight convert: W (K x N, f32, row-major) -> Wt (N x K, bf16, row-major)
// ---------------------------------------------------------------------------
__global__ __launch_bounds__(256)
void k_transpose_bf16(const float* __restrict__ W, bf16_t* __restrict__ Wt, int K, int N) {
    __shared__ float tile[32][33];
    int kb = blockIdx.y * 32, nb = blockIdx.x * 32;
    int tx = threadIdx.x & 31, ty = threadIdx.x >> 5;
    #pragma unroll
    for (int i = ty; i < 32; i += 8) {
        int k = kb + i, n = nb + tx;
        tile[i][tx] = (k < K && n < N) ? W[(size_t)k * N + n] : 0.f;
    }
    __syncthreads();
    #pragma unroll
    for (int i = ty; i < 32; i += 8) {
        int n = nb + i, k = kb + tx;
        if (n < N && k < K) Wt[(size_t)n * K + k] = (bf16_t)tile[tx][i];
    }
}

// ---------------------------------------------------------------------------
// LayerNorm (f32 in) -> bf16 out.  One block per row.
// ---------------------------------------------------------------------------
__global__ __launch_bounds__(256)
void k_layernorm(const float* __restrict__ x, const float* __restrict__ g,
                 const float* __restrict__ b, bf16_t* __restrict__ out, int D) {
    const float* row = x + (size_t)blockIdx.x * D;
    float s = 0.f, s2 = 0.f;
    for (int d = threadIdx.x; d < D; d += 256) { float v = row[d]; s += v; s2 += v * v; }
    #pragma unroll
    for (int off = 16; off; off >>= 1) { s += __shfl_down(s, off); s2 += __shfl_down(s2, off); }
    __shared__ float ws[8], ws2[8];
    int wave = threadIdx.x >> 5, lane = threadIdx.x & 31;
    if (lane == 0) { ws[wave] = s; ws2[wave] = s2; }
    __syncthreads();
    if (threadIdx.x == 0) {
        float a = 0.f, a2 = 0.f;
        #pragma unroll
        for (int i = 0; i < 8; i++) { a += ws[i]; a2 += ws2[i]; }
        ws[0] = a; ws2[0] = a2;
    }
    __syncthreads();
    float mean = ws[0] / D;
    float var  = ws2[0] / D - mean * mean;
    float rstd = rsqrtf(var + 1e-5f);
    bf16_t* orow = out + (size_t)blockIdx.x * D;
    for (int d = threadIdx.x; d < D; d += 256)
        orow[d] = (bf16_t)((row[d] - mean) * rstd * g[d] + b[d]);
}

// ---------------------------------------------------------------------------
// Generic bf16 WMMA GEMM:  C[M,N] = A[M,K] * Wt[N,K]^T  (+bias)(+gelu)(+resid)
// Block tile 128x128, K-step 32, 8 waves: each wave 32(M) x 64(N).
// Async global->LDS double-buffered pipeline when available.
// ---------------------------------------------------------------------------
#define GEMM_LDK 40   // padded LDS K-stride (80B: 16B aligned, conflict-free)

__global__ __launch_bounds__(256)
void k_gemm_bf16(const bf16_t* __restrict__ A, const bf16_t* __restrict__ Wt,
                 const float* __restrict__ bias, const float* __restrict__ resid,
                 float* __restrict__ outF, bf16_t* __restrict__ outB,
                 int M, int N, int K, int act) {
    __shared__ bf16_t As[2][128][GEMM_LDK];
    __shared__ bf16_t Bs[2][128][GEMM_LDK];
    const int tid = threadIdx.x;
    const int wave = tid >> 5, lane = tid & 31;
    const int m0 = blockIdx.y * 128, n0 = blockIdx.x * 128;
    const int wm = (wave & 3) * 32, wn = (wave >> 2) * 64;
    const int lm = lane & 15, kh = lane >> 4;

    v8f acc[2][4];
    #pragma unroll
    for (int i = 0; i < 2; i++)
        #pragma unroll
        for (int j = 0; j < 4; j++)
            #pragma unroll
            for (int v = 0; v < 8; v++) acc[i][j][v] = 0.f;

    const int r = tid >> 1, half = tid & 1;
    const bool bValid = (n0 + r < N);

#if HAS_ASYNC_LDS
    if (!bValid) {           // zero-fill OOB weight rows once (both buffers)
        u32x4 z = {0, 0, 0, 0};
        #pragma unroll
        for (int bufi = 0; bufi < 2; bufi++) {
            *(u32x4*)&Bs[bufi][r][half * 16]     = z;
            *(u32x4*)&Bs[bufi][r][half * 16 + 8] = z;
        }
    }
    {   // prime the pipeline: tile 0 -> buffer 0
        const bf16_t* ga = A + (size_t)(m0 + r) * K + half * 16;
        async_copy16(&As[0][r][half * 16], ga);
        async_copy16(&As[0][r][half * 16 + 8], ga + 8);
        if (bValid) {
            const bf16_t* gb = Wt + (size_t)(n0 + r) * K + half * 16;
            async_copy16(&Bs[0][r][half * 16], gb);
            async_copy16(&Bs[0][r][half * 16 + 8], gb + 8);
        }
    }
    int cur = 0;
    for (int k0 = 0; k0 < K; k0 += 32) {
        async_wait0();
        __syncthreads();
        if (k0 + 32 < K) {   // fetch next tile into the other buffer
            const bf16_t* ga = A + (size_t)(m0 + r) * K + k0 + 32 + half * 16;
            async_copy16(&As[cur ^ 1][r][half * 16], ga);
            async_copy16(&As[cur ^ 1][r][half * 16 + 8], ga + 8);
            if (bValid) {
                const bf16_t* gb = Wt + (size_t)(n0 + r) * K + k0 + 32 + half * 16;
                async_copy16(&Bs[cur ^ 1][r][half * 16], gb);
                async_copy16(&Bs[cur ^ 1][r][half * 16 + 8], gb + 8);
            }
        }
        if (k0 + 64 < K)
            __builtin_prefetch(A + (size_t)(m0 + r) * K + k0 + 64 + half * 16, 0, 3);

        AFrag a[2], bfr[4];
        #pragma unroll
        for (int i = 0; i < 2; i++) {
            const bf16_t* p = &As[cur][wm + i * 16 + lm][kh * 8];
            a[i].q[0] = *(const u32x4*)p;
            a[i].q[1] = *(const u32x4*)(p + 16);
        }
        #pragma unroll
        for (int j = 0; j < 4; j++) {
            const bf16_t* p = &Bs[cur][wn + j * 16 + lm][kh * 8];
            bfr[j].q[0] = *(const u32x4*)p;
            bfr[j].q[1] = *(const u32x4*)(p + 16);
        }
        #pragma unroll
        for (int i = 0; i < 2; i++)
            #pragma unroll
            for (int j = 0; j < 4; j++)
                acc[i][j] = __builtin_amdgcn_wmma_f32_16x16x32_bf16(
                    false, a[i].v, false, bfr[j].v, (short)0, acc[i][j], false, false);
        cur ^= 1;
    }
#else
    for (int k0 = 0; k0 < K; k0 += 32) {
        {
            const bf16_t* gp = A + (size_t)(m0 + r) * K + k0 + half * 16;
            *(u32x4*)&As[0][r][half * 16]     = *(const u32x4*)gp;
            *(u32x4*)&As[0][r][half * 16 + 8] = *(const u32x4*)(gp + 8);
        }
        {
            u32x4 z0 = {0, 0, 0, 0}, z1 = {0, 0, 0, 0};
            if (bValid) {
                const bf16_t* gp = Wt + (size_t)(n0 + r) * K + k0 + half * 16;
                z0 = *(const u32x4*)gp; z1 = *(const u32x4*)(gp + 8);
            }
            *(u32x4*)&Bs[0][r][half * 16]     = z0;
            *(u32x4*)&Bs[0][r][half * 16 + 8] = z1;
        }
        if (k0 + 32 < K)
            __builtin_prefetch(A + (size_t)(m0 + r) * K + k0 + 32 + half * 16, 0, 3);
        __syncthreads();
        AFrag a[2], bfr[4];
        #pragma unroll
        for (int i = 0; i < 2; i++) {
            const bf16_t* p = &As[0][wm + i * 16 + lm][kh * 8];
            a[i].q[0] = *(const u32x4*)p;
            a[i].q[1] = *(const u32x4*)(p + 16);
        }
        #pragma unroll
        for (int j = 0; j < 4; j++) {
            const bf16_t* p = &Bs[0][wn + j * 16 + lm][kh * 8];
            bfr[j].q[0] = *(const u32x4*)p;
            bfr[j].q[1] = *(const u32x4*)(p + 16);
        }
        #pragma unroll
        for (int i = 0; i < 2; i++)
            #pragma unroll
            for (int j = 0; j < 4; j++)
                acc[i][j] = __builtin_amdgcn_wmma_f32_16x16x32_bf16(
                    false, a[i].v, false, bfr[j].v, (short)0, acc[i][j], false, false);
        __syncthreads();
    }
#endif

    // Epilogue
    #pragma unroll
    for (int i = 0; i < 2; i++)
        #pragma unroll
        for (int j = 0; j < 4; j++) {
            int n = n0 + wn + j * 16 + lm;
            if (n >= N) continue;
            float bv = bias ? bias[n] : 0.f;
            #pragma unroll
            for (int v = 0; v < 8; v++) {
                int m = m0 + wm + i * 16 + v + 8 * kh;
                float c = acc[i][j][v] + bv;
                if (act == 1) c = 0.5f * c * (1.f + erff(c * 0.70710678118f));
                size_t idx = (size_t)m * N + n;
                if (resid) c += resid[idx];
                if (outF) outF[idx] = c;
                if (outB) outB[idx] = (bf16_t)c;
            }
        }
}

// ---------------------------------------------------------------------------
// Flash-style causal attention.  qkv bf16 [(b,t), 3, H, 64].
// grid (T/128, B*H), 256 threads (8 waves, 16 q-rows each).
// ---------------------------------------------------------------------------
__global__ __launch_bounds__(256)
void k_attention(const bf16_t* __restrict__ qkv, float* __restrict__ seqout,
                 int B, int T, int H) {
    const int qb = blockIdx.x;
    const int bh = blockIdx.y;
    const int h = bh % H, b = bh / H;
    const int tid = threadIdx.x, wave = tid >> 5, lane = tid & 31;
    const int lm = lane & 15, kh = lane >> 4;

    __shared__ bf16_t Qs[128][72];
    __shared__ bf16_t Ks[128][72];
    __shared__ bf16_t Vt[64][136];     // V transposed: [dh][s]
    __shared__ bf16_t Ps[8][16][40];   // per-wave P staging (32-wide K chunk)

    const size_t strideT = 3 * 1024;
    const bf16_t* base = qkv + (size_t)b * T * strideT + (size_t)h * 64;

    // Load Q tile (128 x 64)
    for (int c8 = tid; c8 < 128 * 8; c8 += 256) {
        int rr = c8 >> 3, cc = (c8 & 7) * 8;
        const bf16_t* gq = base + (size_t)(qb * 128 + rr) * strideT + cc;
#if HAS_ASYNC_LDS
        async_copy16(&Qs[rr][cc], gq);
#else
        *(u32x4*)&Qs[rr][cc] = *(const u32x4*)gq;
#endif
    }

    float rowm[8], rowl[8];
    v8f oacc[4];
    #pragma unroll
    for (int v = 0; v < 8; v++) { rowm[v] = -INFINITY; rowl[v] = 0.f; }
    #pragma unroll
    for (int jt = 0; jt < 4; jt++)
        #pragma unroll
        for (int v = 0; v < 8; v++) oacc[jt][v] = 0.f;

    for (int sb = 0; sb <= qb; sb++) {
        __syncthreads();   // protect Ks/Vt reuse across iterations
        for (int c8 = tid; c8 < 128 * 8; c8 += 256) {
            int rr = c8 >> 3, cc = (c8 & 7) * 8;
            const bf16_t* gk = base + (size_t)(sb * 128 + rr) * strideT + 1024 + cc;
#if HAS_ASYNC_LDS
            async_copy16(&Ks[rr][cc], gk);
#else
            *(u32x4*)&Ks[rr][cc] = *(const u32x4*)gk;
#endif
        }
        for (int c8 = tid; c8 < 128 * 8; c8 += 256) {
            int rr = c8 >> 3, cc = (c8 & 7) * 8;
            union { u32x4 u; bf16_t e[8]; } vv;
            vv.u = *(const u32x4*)(base + (size_t)(sb * 128 + rr) * strideT + 2048 + cc);
            #pragma unroll
            for (int i = 0; i < 8; i++) Vt[cc + i][rr] = vv.e[i];
        }
        async_wait0();
        __syncthreads();

        // ---- scores: S = Q * K^T ----
        AFrag aq[2];
        #pragma unroll
        for (int kk = 0; kk < 2; kk++) {
            const bf16_t* p = &Qs[wave * 16 + lm][kk * 32 + kh * 8];
            aq[kk].q[0] = *(const u32x4*)p;
            aq[kk].q[1] = *(const u32x4*)(p + 16);
        }
        v8f sacc[8];
        #pragma unroll
        for (int j = 0; j < 8; j++)
            #pragma unroll
            for (int v = 0; v < 8; v++) sacc[j][v] = 0.f;
        #pragma unroll
        for (int j = 0; j < 8; j++)
            #pragma unroll
            for (int kk = 0; kk < 2; kk++) {
                AFrag bk;
                const bf16_t* p = &Ks[j * 16 + lm][kk * 32 + kh * 8];
                bk.q[0] = *(const u32x4*)p;
                bk.q[1] = *(const u32x4*)(p + 16);
                sacc[j] = __builtin_amdgcn_wmma_f32_16x16x32_bf16(
                    false, aq[kk].v, false, bk.v, (short)0, sacc[j], false, false);
            }

        // ---- causal mask + online softmax (wave32: 16-lane row reductions) ----
        #pragma unroll
        for (int v = 0; v < 8; v++) {
            int trow = qb * 128 + wave * 16 + v + 8 * kh;
            float rm = -INFINITY;
            #pragma unroll
            for (int j = 0; j < 8; j++) {
                int col = sb * 128 + j * 16 + lm;
                float s = (col <= trow) ? sacc[j][v] * 0.125f : -INFINITY;
                sacc[j][v] = s;
                rm = fmaxf(rm, s);
            }
            #pragma unroll
            for (int msk = 1; msk < 16; msk <<= 1) rm = fmaxf(rm, __shfl_xor(rm, msk));
            float mnew = fmaxf(rowm[v], rm);
            float corr = expf(rowm[v] - mnew);
            rowm[v] = mnew;
            float rs = 0.f;
            #pragma unroll
            for (int j = 0; j < 8; j++) {
                float p = expf(sacc[j][v] - mnew);
                sacc[j][v] = p;
                rs += p;
            }
            #pragma unroll
            for (int msk = 1; msk < 16; msk <<= 1) rs += __shfl_xor(rs, msk);
            rowl[v] = rowl[v] * corr + rs;
            #pragma unroll
            for (int jt = 0; jt < 4; jt++) oacc[jt][v] = oacc[jt][v] * corr;
        }

        // ---- O += P * V  (stage P through LDS: C-layout -> A-layout) ----
        #pragma unroll
        for (int kk = 0; kk < 4; kk++) {
            #pragma unroll
            for (int jj = 0; jj < 2; jj++) {
                int j = kk * 2 + jj;
                #pragma unroll
                for (int v = 0; v < 8; v++)
                    Ps[wave][v + 8 * kh][jj * 16 + lm] = (bf16_t)sacc[j][v];
            }
            AFrag ap;
            const bf16_t* p = &Ps[wave][lm][kh * 8];
            ap.q[0] = *(const u32x4*)p;
            ap.q[1] = *(const u32x4*)(p + 16);
            #pragma unroll
            for (int jt = 0; jt < 4; jt++) {
                AFrag bv;
                const bf16_t* q = &Vt[jt * 16 + lm][kk * 32 + kh * 8];
                bv.q[0] = *(const u32x4*)q;
                bv.q[1] = *(const u32x4*)(q + 16);
                oacc[jt] = __builtin_amdgcn_wmma_f32_16x16x32_bf16(
                    false, ap.v, false, bv.v, (short)0, oacc[jt], false, false);
            }
        }
    }

    float* out = seqout + (size_t)b * T * 1024 + (size_t)h * 64;
    #pragma unroll
    for (int jt = 0; jt < 4; jt++)
        #pragma unroll
        for (int v = 0; v < 8; v++) {
            int trow = qb * 128 + wave * 16 + v + 8 * kh;
            int col  = jt * 16 + lm;
            out[(size_t)trow * 1024 + col] = oacc[jt][v] / rowl[v];
        }
}

// ---------------------------------------------------------------------------
// Plucker lines + J6 rotation.  One thread per (b,t,h).
// ---------------------------------------------------------------------------
__device__ inline void exterior4(const float* a1, const float* a2, float* L) {
    L[0] = a1[0] * a2[1] - a1[1] * a2[0];
    L[1] = a1[0] * a2[2] - a1[2] * a2[0];
    L[2] = a1[0] * a2[3] - a1[3] * a2[0];
    L[3] = a1[1] * a2[2] - a1[2] * a2[1];
    L[4] = a1[1] * a2[3] - a1[3] * a2[1];
    L[5] = a1[2] * a2[3] - a1[3] * a2[2];
    float n2 = 0.f;
    #pragma unroll
    for (int i = 0; i < 6; i++) n2 += L[i] * L[i];
    float inv = 1.f / fmaxf(sqrtf(n2), 1e-12f);
    #pragma unroll
    for (int i = 0; i < 6; i++) L[i] *= inv;
}

__global__ __launch_bounds__(256)
void k_lines(const float* __restrict__ p1w, const float* __restrict__ p2w,
             const float* __restrict__ p1r, const float* __restrict__ p2r,
             float* __restrict__ Jw, float* __restrict__ Rd, int B, int T, int H) {
    int idx = blockIdx.x * 256 + threadIdx.x;
    if (idx >= B * T * H) return;
    int h = idx % H, t = (idx / H) % T, b = idx / (H * T);
    size_t row = (size_t)b * T + t;
    float a1[4], a2[4], L[6];
    #pragma unroll
    for (int c = 0; c < 4; c++) {
        a1[c] = (t > 0) ? p1w[(row - 1) * 64 + h * 4 + c] : 0.f;
        a2[c] = p2w[row * 64 + h * 4 + c];
    }
    exterior4(a1, a2, L);
    float* jw = Jw + ((size_t)(b * H + h) * T + t) * 6;
    jw[0] = L[5]; jw[1] = -L[4]; jw[2] = L[3]; jw[3] = L[2]; jw[4] = -L[1]; jw[5] = L[0];
    #pragma unroll
    for (int c = 0; c < 4; c++) {
        a1[c] = p1r[row * 64 + h * 4 + c];
        a2[c] = p2r[row * 64 + h * 4 + c];
    }
    exterior4(a1, a2, L);
    float* rd = Rd + ((size_t)(b * H + h) * T + t) * 6;
    #pragma unroll
    for (int j = 0; j < 6; j++) rd[j] = L[j];
}

// ---------------------------------------------------------------------------
// Sequential decayed-Gram scan:  M_t = d*(M_{t-1} + Jw_{t-1} Jw_{t-1}^T).
// One wave per (b,h); lane j<6 owns column j of the 6x6 M.
// ---------------------------------------------------------------------------
__global__ __launch_bounds__(32)
void k_gram_scan(const float* __restrict__ Jw, const float* __restrict__ Rd,
                 const float* __restrict__ glin, const float* __restrict__ decay_logits,
                 const float* __restrict__ iter_mix, const float* __restrict__ mem_scale,
                 float* __restrict__ gh, int B, int T, int H) {
    int bh = blockIdx.x, h = bh % H, b = bh / H;
    int j = threadIdx.x;
    float decay = 1.f / (1.f + expf(-decay_logits[h]));
    float alpha = 1.f / (1.f + expf(-iter_mix[0]));
    float msc = mem_scale[h];
    float Mcol[6] = {0, 0, 0, 0, 0, 0};
    float pJ[6]   = {0, 0, 0, 0, 0, 0};
    const float* Jb = Jw + (size_t)bh * T * 6;
    const float* Rb = Rd + (size_t)bh * T * 6;
    for (int t = 0; t < T; t++) {
        float rd[6], jw[6];
        #pragma unroll
        for (int i = 0; i < 6; i++) { rd[i] = Rb[t * 6 + i]; jw[i] = Jb[t * 6 + i]; }
        float s1 = 0.f, s2 = 0.f;
        if (j < 6) {
            float pj = pJ[j], rdM = 0.f;
            #pragma unroll
            for (int i = 0; i < 6; i++) {
                Mcol[i] = decay * (Mcol[i] + pJ[i] * pj);
                rdM += rd[i] * Mcol[i];
            }
            s1 = rdM * rd[j];
            s2 = rdM * rdM;
        }
        #pragma unroll
        for (int m = 1; m < 8; m <<= 1) { s1 += __shfl_xor(s1, m); s2 += __shfl_xor(s2, m); }
        if (j == 0) {
            float ms = (1.f - alpha) * s1 + alpha * s2;
            float g1 = 1.f / (1.f + expf(-ms * msc));
            float g2 = 1.f / (1.f + expf(-glin[((size_t)b * T + t) * H + h]));
            gh[((size_t)b * T + t) * H + h] = g1 * g2;
        }
        #pragma unroll
        for (int i = 0; i < 6; i++) pJ[i] = jw[i];
    }
}

// ---------------------------------------------------------------------------
// y = seqout + mean_h(gh) * memval  -> bf16 (input to Wout GEMM)
// ---------------------------------------------------------------------------
__global__ __launch_bounds__(256)
void k_combine(const float* __restrict__ seqout, const float* __restrict__ memval,
               const float* __restrict__ gh, bf16_t* __restrict__ y, int D, int H) {
    size_t row = blockIdx.x;
    float g = 0.f;
    #pragma unroll
    for (int h = 0; h < 16; h++) g += gh[row * H + h];
    g *= (1.f / 16.f);
    const float* so = seqout + row * D;
    const float* mv = memval + row * D;
    bf16_t* yr = y + row * D;
    for (int d = threadIdx.x; d < D; d += 256)
        yr[d] = (bf16_t)(so[d] + g * mv[d]);
}

// ---------------------------------------------------------------------------
extern "C" void kernel_launch(void* const* d_in, const int* in_sizes, int n_in,
                              void* d_out, int out_size, void* d_ws, size_t ws_size,
                              hipStream_t stream) {
    const int B = 2, T = 1024, D = 1024, H = 16, DFF = 4096;
    const int M = B * T;

    const float* x            = (const float*)d_in[0];
    const float* ln1_g        = (const float*)d_in[1];
    const float* ln1_b        = (const float*)d_in[2];
    const float* Wqkv         = (const float*)d_in[3];
    const float* bqkv         = (const float*)d_in[4];
    const float* W1w          = (const float*)d_in[5];
    const float* W2w          = (const float*)d_in[6];
    const float* W1r          = (const float*)d_in[7];
    const float* W2r          = (const float*)d_in[8];
    const float* Wmv          = (const float*)d_in[9];
    const float* bmv          = (const float*)d_in[10];
    const float* Wmg          = (const float*)d_in[11];
    const float* bmg          = (const float*)d_in[12];
    const float* mem_scale    = (const float*)d_in[13];
    const float* iter_mix     = (const float*)d_in[14];
    const float* decay_logits = (const float*)d_in[15];
    const float* Wout         = (const float*)d_in[16];
    const float* bout         = (const float*)d_in[17];
    const float* ln2_g        = (const float*)d_in[18];
    const float* ln2_b        = (const float*)d_in[19];
    const float* Wfc          = (const float*)d_in[20];
    const float* bfc          = (const float*)d_in[21];
    const float* Wproj        = (const float*)d_in[22];
    const float* bproj        = (const float*)d_in[23];

    char* ws = (char*)d_ws;
    size_t off = 0;
    auto alloc = [&](size_t bytes) -> void* {
        void* p = ws + off;
        off += (bytes + 255) & ~(size_t)255;
        return p;
    };

    bf16_t* Wqkv_t  = (bf16_t*)alloc((size_t)3 * D * D * 2);
    bf16_t* W1w_t   = (bf16_t*)alloc((size_t)64 * D * 2);
    bf16_t* W2w_t   = (bf16_t*)alloc((size_t)64 * D * 2);
    bf16_t* W1r_t   = (bf16_t*)alloc((size_t)64 * D * 2);
    bf16_t* W2r_t   = (bf16_t*)alloc((size_t)64 * D * 2);
    bf16_t* Wmg_t   = (bf16_t*)alloc((size_t)H * D * 2);
    bf16_t* Wmv_t   = (bf16_t*)alloc((size_t)D * D * 2);
    bf16_t* Wout_t  = (bf16_t*)alloc((size_t)D * D * 2);
    bf16_t* Wfc_t   = (bf16_t*)alloc((size_t)DFF * D * 2);
    bf16_t* Wproj_t = (bf16_t*)alloc((size_t)D * DFF * 2);
    bf16_t* h_bf    = (bf16_t*)alloc((size_t)M * D * 2);
    bf16_t* qkv_bf  = (bf16_t*)alloc((size_t)M * 3 * D * 2);
    float*  p1w     = (float*)alloc((size_t)M * 64 * 4);
    float*  p2w     = (float*)alloc((size_t)M * 64 * 4);
    float*  p1r     = (float*)alloc((size_t)M * 64 * 4);
    float*  p2r     = (float*)alloc((size_t)M * 64 * 4);
    float*  glin    = (float*)alloc((size_t)M * H * 4);
    float*  Jwb     = (float*)alloc((size_t)B * H * T * 6 * 4);
    float*  Rdb     = (float*)alloc((size_t)B * H * T * 6 * 4);
    float*  ghb     = (float*)alloc((size_t)M * H * 4);
    float*  memval  = (float*)alloc((size_t)M * D * 4);
    float*  seqout  = (float*)alloc((size_t)M * D * 4);
    bf16_t* y_bf    = (bf16_t*)alloc((size_t)M * D * 2);
    float*  x1      = (float*)alloc((size_t)M * D * 4);
    bf16_t* h2_bf   = (bf16_t*)alloc((size_t)M * D * 2);
    bf16_t* a_bf    = (bf16_t*)alloc((size_t)M * DFF * 2);

    auto tr = [&](const float* W, bf16_t* Wt, int K, int N) {
        dim3 g((N + 31) / 32, (K + 31) / 32);
        k_transpose_bf16<<<g, 256, 0, stream>>>(W, Wt, K, N);
    };
    auto gemm = [&](const bf16_t* A, const bf16_t* Wt, const float* bias,
                    const float* resid, float* oF, bf16_t* oB, int m, int n, int k, int act) {
        dim3 g((n + 127) / 128, (m + 127) / 128);
        k_gemm_bf16<<<g, 256, 0, stream>>>(A, Wt, bias, resid, oF, oB, m, n, k, act);
    };

    // 1) weights -> bf16 transposed
    tr(Wqkv, Wqkv_t, D, 3 * D);
    tr(W1w, W1w_t, D, 64);
    tr(W2w, W2w_t, D, 64);
    tr(W1r, W1r_t, D, 64);
    tr(W2r, W2r_t, D, 64);
    tr(Wmg, Wmg_t, D, H);
    tr(Wmv, Wmv_t, D, D);
    tr(Wout, Wout_t, D, D);
    tr(Wfc, Wfc_t, D, DFF);
    tr(Wproj, Wproj_t, DFF, D);

    // 2) ln1
    k_layernorm<<<M, 256, 0, stream>>>(x, ln1_g, ln1_b, h_bf, D);

    // 3) projections
    gemm(h_bf, Wqkv_t, bqkv, nullptr, nullptr, qkv_bf, M, 3 * D, D, 0);
    gemm(h_bf, W1w_t, nullptr, nullptr, p1w, nullptr, M, 64, D, 0);
    gemm(h_bf, W2w_t, nullptr, nullptr, p2w, nullptr, M, 64, D, 0);
    gemm(h_bf, W1r_t, nullptr, nullptr, p1r, nullptr, M, 64, D, 0);
    gemm(h_bf, W2r_t, nullptr, nullptr, p2r, nullptr, M, 64, D, 0);
    gemm(h_bf, Wmg_t, bmg, nullptr, glin, nullptr, M, H, D, 0);
    gemm(h_bf, Wmv_t, bmv, nullptr, memval, nullptr, M, D, D, 0);

    // 4) attention
    k_attention<<<dim3(T / 128, B * H), 256, 0, stream>>>(qkv_bf, seqout, B, T, H);

    // 5) Gram memory branch
    k_lines<<<(B * T * H + 255) / 256, 256, 0, stream>>>(p1w, p2w, p1r, p2r, Jwb, Rdb, B, T, H);
    k_gram_scan<<<B * H, 32, 0, stream>>>(Jwb, Rdb, glin, decay_logits, iter_mix,
                                          mem_scale, ghb, B, T, H);
    k_combine<<<M, 256, 0, stream>>>(seqout, memval, ghb, y_bf, D, H);

    // 6) output projection + residual
    gemm(y_bf, Wout_t, bout, x, x1, nullptr, M, D, D, 0);

    // 7) MLP
    k_layernorm<<<M, 256, 0, stream>>>(x1, ln2_g, ln2_b, h2_bf, D);
    gemm(h2_bf, Wfc_t, bfc, nullptr, nullptr, a_bf, M, DFF, D, 1);
    gemm(a_bf, Wproj_t, bproj, x1, (float*)d_out, nullptr, M, D, DFF, 0);
}
